// LinearXNOR_86260123173274
// MI455X (gfx1250) — 8-bit matrix operand order repaired and checked by reading, compiled, not run on hardware
//
#include <hip/hip_runtime.h>
#include <cstdint>
#include <cstddef>

#define IN_F   4096
#define OUT_F  16384
#define NTOK   8192

typedef __attribute__((ext_vector_type(16))) int   v16i;
typedef __attribute__((ext_vector_type(8)))  float v8f;
typedef __attribute__((ext_vector_type(4)))  float v4f;
typedef __attribute__((ext_vector_type(4)))  unsigned int v4u;

__global__ void zero_scalar_kernel(float* p) {
    if (threadIdx.x == 0 && blockIdx.x == 0) p[0] = 0.0f;
}

__global__ __launch_bounds__(256) void abs_sum_kernel(const float* __restrict__ w,
                                                      float* __restrict__ out,
                                                      long n4) {
    __shared__ float red[256];
    const v4f* src = (const v4f*)w;
    float s = 0.0f;
    for (long i = (long)blockIdx.x * blockDim.x + threadIdx.x; i < n4;
         i += (long)gridDim.x * blockDim.x) {
        v4f v = __builtin_nontemporal_load(src + i);
        s += fabsf(v.x) + fabsf(v.y) + fabsf(v.z) + fabsf(v.w);
    }
    red[threadIdx.x] = s;
    __syncthreads();
    for (int o = 128; o > 0; o >>= 1) {
        if (threadIdx.x < o) red[threadIdx.x] += red[threadIdx.x + o];
        __syncthreads();
    }
    if (threadIdx.x == 0) atomicAdd(out, red[0]);
}

__device__ __forceinline__ unsigned sgn8(float v) {
    return v > 0.0f ? 0x38u : (v < 0.0f ? 0xB8u : 0x00u);
}

__global__ __launch_bounds__(256) void binarize_fp8_kernel(const float* __restrict__ src,
                                                           v4u* __restrict__ dst,
                                                           long n16) {
    long i = (long)blockIdx.x * blockDim.x + threadIdx.x;
    if (i >= n16) return;
    const v4f* s = (const v4f*)src + i * 4;
    v4f f0 = __builtin_nontemporal_load(s + 0);
    v4f f1 = __builtin_nontemporal_load(s + 1);
    v4f f2 = __builtin_nontemporal_load(s + 2);
    v4f f3 = __builtin_nontemporal_load(s + 3);
    v4u o;
    o.x = sgn8(f0.x) | (sgn8(f0.y) << 8) | (sgn8(f0.z) << 16) | (sgn8(f0.w) << 24);
    o.y = sgn8(f1.x) | (sgn8(f1.y) << 8) | (sgn8(f1.z) << 16) | (sgn8(f1.w) << 24);
    o.z = sgn8(f2.x) | (sgn8(f2.y) << 8) | (sgn8(f2.z) << 16) | (sgn8(f2.w) << 24);
    o.w = sgn8(f3.x) | (sgn8(f3.y) << 8) | (sgn8(f3.z) << 16) | (sgn8(f3.w) << 24);
    dst[i] = o;
}

__device__ __forceinline__ void async_copy_b128(unsigned lds_off, const unsigned char* g) {
    asm volatile("global_load_async_to_lds_b128 %0, %1, off"
                 :: "v"(lds_off), "v"((unsigned long long)(uintptr_t)g)
                 : "memory");
}
__device__ __forceinline__ void wait_asynccnt0() {
    asm volatile("s_wait_asynccnt 0x0" ::: "memory");
}
__device__ __forceinline__ unsigned lds_off32(const void* p) {
    return (unsigned)(uintptr_t)p;
}

#define LDA 144

__global__ __launch_bounds__(256) void xnor_gemm_kernel(
    const unsigned char* __restrict__ A8,
    const unsigned char* __restrict__ B8,
    const float* __restrict__ bias,
    const float* __restrict__ alphaSum,
    float* __restrict__ out)
{
    __shared__ __align__(16) unsigned char As[2][128 * LDA];
    __shared__ __align__(16) unsigned char Bs[2][128 * LDA];

    const int n0 = blockIdx.x * 128;
    const int m0 = blockIdx.y * 128;
    const int tid  = threadIdx.x;
    const int lane = tid & 31;
    const int wid  = tid >> 5;
    const int wm = wid & 3;
    const int wn = wid >> 2;
    const int lrow  = lane & 15;
    const int lhalf = lane >> 4;

    v8f acc[2][4] = {};

#pragma unroll
    for (int i = 0; i < 4; ++i) {
        int c   = tid + i * 256;
        int row = c >> 3;
        int col = (c & 7) * 16;
        async_copy_b128(lds_off32(&As[0][row * LDA + col]),
                        A8 + (size_t)(m0 + row) * IN_F + col);
        async_copy_b128(lds_off32(&Bs[0][row * LDA + col]),
                        B8 + (size_t)(n0 + row) * IN_F + col);
    }
    wait_asynccnt0();
    __syncthreads();

    int buf = 0;
    for (int k0 = 0; k0 < IN_F; k0 += 128) {
        const bool has_next = (k0 + 128 < IN_F);
        if (has_next) {
            const int kn = k0 + 128;
#pragma unroll
            for (int i = 0; i < 4; ++i) {
                int c   = tid + i * 256;
                int row = c >> 3;
                int col = (c & 7) * 16;
                async_copy_b128(lds_off32(&As[buf ^ 1][row * LDA + col]),
                                A8 + (size_t)(m0 + row) * IN_F + kn + col);
                async_copy_b128(lds_off32(&Bs[buf ^ 1][row * LDA + col]),
                                B8 + (size_t)(n0 + row) * IN_F + kn + col);
            }
        }
        if (k0 + 256 < IN_F) {
            __builtin_prefetch(A8 + (size_t)(m0 + (tid >> 1)) * IN_F + k0 + 256 + (tid & 1) * 64, 0, 1);
            __builtin_prefetch(B8 + (size_t)(n0 + (tid >> 1)) * IN_F + k0 + 256 + (tid & 1) * 64, 0, 1);
        }

        v16i af[2];
#pragma unroll
        for (int mi = 0; mi < 2; ++mi) {
            const unsigned char* rp = &As[buf][(wm * 32 + mi * 16 + lrow) * LDA];
#pragma unroll
            for (int blk = 0; blk < 2; ++blk) {
#pragma unroll
                for (int j = 0; j < 4; ++j) {
                    unsigned long long d =
                        *(const unsigned long long*)(rp + blk * 64 + (j >> 1) * 32 + lhalf * 16 + (j & 1) * 8);
                    af[mi][blk * 8 + 2 * j]     = (int)(unsigned)d;
                    af[mi][blk * 8 + 2 * j + 1] = (int)(d >> 32);
                }
            }
        }
        v16i bf[4];
#pragma unroll
        for (int ni = 0; ni < 4; ++ni) {
            const unsigned char* cp = &Bs[buf][(wn * 64 + ni * 16 + lrow) * LDA];
#pragma unroll
            for (int g = 0; g < 4; ++g) {
                v4u d = *(const v4u*)(cp + g * 32 + lhalf * 16);
                bf[ni][4 * g + 0] = (int)d.x;
                bf[ni][4 * g + 1] = (int)d.y;
                bf[ni][4 * g + 2] = (int)d.z;
                bf[ni][4 * g + 3] = (int)d.w;
            }
        }
#pragma unroll
        for (int mi = 0; mi < 2; ++mi)
#pragma unroll
            for (int ni = 0; ni < 4; ++ni)
                acc[mi][ni] = __builtin_amdgcn_wmma_f32_16x16x128_fp8_fp8(
                    af[mi], bf[ni], (short)0, acc[mi][ni], false, false);

        if (has_next) {
            wait_asynccnt0();
            __syncthreads();
            buf ^= 1;
        }
    }

    const float alpha = alphaSum[0] * (1.0f / ((float)OUT_F * (float)IN_F));
#pragma unroll
    for (int mi = 0; mi < 2; ++mi) {
#pragma unroll
        for (int ni = 0; ni < 4; ++ni) {
            int gcol = n0 + wn * 64 + ni * 16 + lrow;
            float bv = bias[gcol];
            int growb = m0 + wm * 32 + mi * 16 + lhalf * 8;
#pragma unroll
            for (int r = 0; r < 8; ++r) {
                __builtin_nontemporal_store(acc[mi][ni][r] * alpha + bv,
                                            out + (size_t)(growb + r) * OUT_F + gcol);
            }
        }
    }
}

extern "C" void kernel_launch(void* const* d_in, const int* in_sizes, int n_in,
                              void* d_out, int out_size, void* d_ws, size_t ws_size,
                              hipStream_t stream) {
    const float* x    = (const float*)d_in[0];
    const float* w    = (const float*)d_in[1];
    const float* bias = (const float*)d_in[2];
    float* out = (float*)d_out;

    float*         alphaSum = (float*)d_ws;
    unsigned char* x8 = (unsigned char*)d_ws + 256;
    unsigned char* w8 = x8 + (size_t)NTOK * IN_F;

    zero_scalar_kernel<<<1, 1, 0, stream>>>(alphaSum);
    abs_sum_kernel<<<2048, 256, 0, stream>>>(w, alphaSum, (long)OUT_F * IN_F / 4);

    const long x16 = (long)NTOK * IN_F / 16;
    const long w16 = (long)OUT_F * IN_F / 16;
    binarize_fp8_kernel<<<(unsigned)(x16 / 256), 256, 0, stream>>>(x, (v4u*)x8, x16);
    binarize_fp8_kernel<<<(unsigned)(w16 / 256), 256, 0, stream>>>(w, (v4u*)w8, w16);

    dim3 grid(OUT_F / 128, NTOK / 128);
    xnor_gemm_kernel<<<grid, 256, 0, stream>>>(x8, w8, bias, alphaSum, out);
}
